// ResNet_quant_21174188769309
// MI455X (gfx1250) — 8-bit matrix operand order repaired and checked by reading, compiled, not run on hardware
//
#include <hip/hip_runtime.h>
#include <stdint.h>

typedef int v8i __attribute__((ext_vector_type(8)));

__device__ __forceinline__ int8_t dsgn(float v) {
    return v > 0.f ? (int8_t)1 : (v < 0.f ? (int8_t)-1 : (int8_t)0);
}

__global__ void kZero16(int8_t* __restrict__ p, int count16) {
    int idx = blockIdx.x * blockDim.x + threadIdx.x;
    if (idx >= count16) return;
    ((int4*)p)[idx] = make_int4(0, 0, 0, 0);
}

__global__ void kQuantW(const float* __restrict__ w, int8_t* __restrict__ out,
                        int O, int I, int K) {
    int idx = blockIdx.x * blockDim.x + threadIdx.x;
    int T = K * K;
    int total = O * I * T;
    if (idx >= total) return;
    int ci = idx % I;
    int t  = (idx / I) % T;
    int co = idx / (I * T);
    int ky = t / K, kx = t % K;
    float v = w[((co * I + ci) * K + ky) * K + kx];
    out[idx] = dsgn(v);
}

__global__ void kStem(const float* __restrict__ x, const int8_t* __restrict__ w8,
                      float* __restrict__ out) {
    int idx = blockIdx.x * blockDim.x + threadIdx.x;
    if (idx >= 131072 * 64) return;
    int co = idx & 63;
    int n  = idx >> 6;
    int img = n >> 10;
    int rem = n & 1023;
    int oy = rem >> 5, ox = rem & 31;
    float acc = 0.f;
    for (int ky = 0; ky < 3; ++ky) {
        int iy = oy + ky - 1;
        if (iy < 0 || iy >= 32) continue;
        for (int kx = 0; kx < 3; ++kx) {
            int ix = ox + kx - 1;
            if (ix < 0 || ix >= 32) continue;
            for (int ci = 0; ci < 3; ++ci) {
                acc += x[((img * 3 + ci) * 32 + iy) * 32 + ix] *
                       (float)w8[(co * 9 + ky * 3 + kx) * 3 + ci];
            }
        }
    }
    out[(size_t)n * 64 + co] = acc;
}

__global__ void __launch_bounds__(32)
kBinConvWmma(const int8_t* __restrict__ act, const int8_t* __restrict__ w8,
             float* __restrict__ out,
             int Cin, int Cout, int Hin, int Win, int Hout, int Wout,
             int stride, int pad, int khw) {
    const int l  = threadIdx.x;
    const int lm = l & 15;
    const int h  = l >> 4;
    const int ntile = blockIdx.x;
    const int mtile = blockIdx.y;

    const int HW = Hout * Wout;
    const int Hp = Hin + 2, Wp = Win + 2;

    int simg[4], soy[4], sox[4];
#pragma unroll
    for (int s = 0; s < 4; ++s) {
        int n   = ntile * 64 + s * 16 + lm;
        simg[s] = n / HW;
        int rem = n % HW;
        soy[s]  = rem / Wout;
        sox[s]  = rem % Wout;
    }
    const int co = mtile * 16 + lm;
    const int T  = khw * khw;

    v8i acc0 = {0, 0, 0, 0, 0, 0, 0, 0};
    v8i acc1 = {0, 0, 0, 0, 0, 0, 0, 0};
    v8i acc2 = {0, 0, 0, 0, 0, 0, 0, 0};
    v8i acc3 = {0, 0, 0, 0, 0, 0, 0, 0};

    for (int t = 0; t < T; ++t) {
        const int ky = t / khw, kx = t % khw;
        size_t bbase[4];
#pragma unroll
        for (int s = 0; s < 4; ++s) {
            const int iy = soy[s] * stride + ky - pad + 1;
            const int ix = sox[s] * stride + kx - pad + 1;
            bbase[s] = ((size_t)((simg[s] * Hp + iy) * Wp + ix)) * (size_t)Cin;
        }
        const size_t abase = ((size_t)co * T + t) * (size_t)Cin;

        for (int ck = 0; ck < Cin; ck += 64) {
            v8i af;
            const int8_t* ap = w8 + abase + ck;
#pragma unroll
            for (int i = 0; i < 2; ++i) {
                const int4 q = *(const int4*)(ap + 32 * i + 16 * h);
                af[4 * i]     = q.x;
                af[4 * i + 1] = q.y;
                af[4 * i + 2] = q.z;
                af[4 * i + 3] = q.w;
            }
            v8i bf[4];
#pragma unroll
            for (int s = 0; s < 4; ++s) {
                const int8_t* bp = act + bbase[s] + ck + 16 * h;
                const int4 b0 = *(const int4*)(bp);
                const int4 b1 = *(const int4*)(bp + 32);
                bf[s][0] = b0.x; bf[s][1] = b0.y; bf[s][2] = b0.z; bf[s][3] = b0.w;
                bf[s][4] = b1.x; bf[s][5] = b1.y; bf[s][6] = b1.z; bf[s][7] = b1.w;
            }
            acc0 = __builtin_amdgcn_wmma_i32_16x16x64_iu8(true, af, true, bf[0], acc0, false, false);
            acc1 = __builtin_amdgcn_wmma_i32_16x16x64_iu8(true, af, true, bf[1], acc1, false, false);
            acc2 = __builtin_amdgcn_wmma_i32_16x16x64_iu8(true, af, true, bf[2], acc2, false, false);
            acc3 = __builtin_amdgcn_wmma_i32_16x16x64_iu8(true, af, true, bf[3], acc3, false, false);
        }
    }

    const size_t cobase = (size_t)mtile * 16 + 8 * h;
#pragma unroll
    for (int s = 0; s < 4; ++s) {
        const int n = ntile * 64 + s * 16 + lm;
        float* op = out + (size_t)n * Cout + cobase;
        v8i a = (s == 0) ? acc0 : (s == 1) ? acc1 : (s == 2) ? acc2 : acc3;
#pragma unroll
        for (int r = 0; r < 8; ++r) op[r] = (float)a[r];
    }
}

__global__ void kBNStats(const float* __restrict__ xf, int C, int Ncol,
                         float* __restrict__ mean, float* __restrict__ rinv) {
    __shared__ float ss[256];
    __shared__ float sq[256];
    const int c = blockIdx.x;
    float s = 0.f, q = 0.f;
    for (int n = threadIdx.x; n < Ncol; n += 256) {
        float v = xf[(size_t)n * C + c];
        s += v; q += v * v;
    }
    ss[threadIdx.x] = s; sq[threadIdx.x] = q;
    __syncthreads();
    for (int st = 128; st > 0; st >>= 1) {
        if (threadIdx.x < st) {
            ss[threadIdx.x] += ss[threadIdx.x + st];
            sq[threadIdx.x] += sq[threadIdx.x + st];
        }
        __syncthreads();
    }
    if (threadIdx.x == 0) {
        float m = ss[0] / (float)Ncol;
        float v = sq[0] / (float)Ncol - m * m;
        mean[c] = m;
        rinv[c] = rsqrtf(v + 1e-5f);
    }
}

__device__ __forceinline__ size_t padOff(int n, int c, int H, int W, int C) {
    int HW  = H * W;
    int img = n / HW;
    int rem = n % HW;
    int oy  = rem / W;
    int ox  = rem % W;
    return ((size_t)((img * (H + 2) + oy + 1) * (W + 2) + ox + 1)) * (size_t)C + c;
}

__global__ void kBNQuantPad(const float* __restrict__ xf, const float* __restrict__ g,
                            const float* __restrict__ b, const float* __restrict__ mean,
                            const float* __restrict__ rinv, int8_t* __restrict__ act,
                            int C, int H, int W, int total) {
    int idx = blockIdx.x * blockDim.x + threadIdx.x;
    if (idx >= total) return;
    int c = idx % C;
    int n = idx / C;
    float y = g[c] * (xf[idx] - mean[c]) * rinv[c] + b[c];
    act[padOff(n, c, H, W, C)] = dsgn(y);
}

__global__ void kBNApply(float* __restrict__ xf, const float* __restrict__ g,
                         const float* __restrict__ b, const float* __restrict__ mean,
                         const float* __restrict__ rinv, int C, int total) {
    int idx = blockIdx.x * blockDim.x + threadIdx.x;
    if (idx >= total) return;
    int c = idx % C;
    xf[idx] = g[c] * (xf[idx] - mean[c]) * rinv[c] + b[c];
}

__global__ void kAddQuantFPad(const float* __restrict__ hh, const float* __restrict__ sc,
                              int8_t* __restrict__ act, int C, int H, int W, int total) {
    int idx = blockIdx.x * blockDim.x + threadIdx.x;
    if (idx >= total) return;
    int c = idx % C;
    int n = idx / C;
    act[padOff(n, c, H, W, C)] = dsgn(hh[idx] + sc[idx]);
}

__global__ void kAddQuantIPad(const float* __restrict__ hh, const int8_t* __restrict__ sc,
                              int8_t* __restrict__ act, int C, int H, int W, int total) {
    int idx = blockIdx.x * blockDim.x + threadIdx.x;
    if (idx >= total) return;
    int c = idx % C;
    int n = idx / C;
    size_t o = padOff(n, c, H, W, C);
    act[o] = dsgn(hh[idx] + (float)sc[o]);
}

__global__ void kFC(const int8_t* __restrict__ act, const float* __restrict__ fcw,
                    const float* __restrict__ fcb, float* __restrict__ out) {
    int idx = blockIdx.x * blockDim.x + threadIdx.x;
    if (idx >= 1280) return;
    int img = idx / 10, k = idx % 10;
    float acc = 0.f;
    for (int c = 0; c < 512; ++c) {
        int s = 0;
        for (int p = 0; p < 16; ++p) {
            int oy = p >> 2, ox = p & 3;
            s += (int)act[((size_t)((img * 6 + oy + 1) * 6 + ox + 1)) * 512 + c];
        }
        float w = fcw[k * 512 + c];
        float sw = w > 0.f ? 1.f : (w < 0.f ? -1.f : 0.f);
        acc += (float)s * sw;
    }
    out[idx] = acc * (1.f / 16.f) + fcb[k];
}

extern "C" void kernel_launch(void* const* d_in, const int* in_sizes, int n_in,
                              void* d_out, int out_size, void* d_ws, size_t ws_size,
                              hipStream_t stream) {
    (void)in_sizes; (void)n_in; (void)out_size; (void)ws_size;

    int pi = 0;
    auto nxtp = [&]() { return (const float*)d_in[pi++]; };

    const float* x      = nxtp();
    const float* conv1w = nxtp();
    const float* bn1g   = nxtp();
    const float* bn1b   = nxtp();

    struct Blk {
        const float *c1w, *g1, *b1, *c2w, *g2, *b2, *scw, *gs, *bs;
        int inp, pl, st;
    };
    Blk blks[8];
    const int planes_arr[4] = {64, 128, 256, 512};
    const int stride_arr[4] = {1, 2, 2, 2};
    int in_p = 64, nb = 0;
    for (int si = 0; si < 4; ++si) {
        for (int bi = 0; bi < 2; ++bi) {
            Blk b;
            b.pl  = planes_arr[si];
            b.st  = (bi == 0) ? stride_arr[si] : 1;
            b.inp = in_p;
            b.c1w = nxtp(); b.g1 = nxtp(); b.b1 = nxtp();
            b.c2w = nxtp(); b.g2 = nxtp(); b.b2 = nxtp();
            if (b.st != 1 || b.inp != b.pl) {
                b.scw = nxtp(); b.gs = nxtp(); b.bs = nxtp();
            } else {
                b.scw = nullptr; b.gs = nullptr; b.bs = nullptr;
            }
            blks[nb++] = b;
            in_p = b.pl;
        }
    }
    const float* fcw = nxtp();
    const float* fcb = nxtp();

    char* ws = (char*)d_ws;
    const size_t SZ_ACT = 10485760ull;
    int8_t* actA = (int8_t*)ws;
    int8_t* actB = (int8_t*)(ws + SZ_ACT);
    int8_t* actH = (int8_t*)(ws + 2 * SZ_ACT);
    float*  F1   = (float*)(ws + 3 * SZ_ACT);
    float*  F2   = (float*)(ws + 3 * SZ_ACT + 33554432ull);
    char*   wq   = ws + 3 * SZ_ACT + 33554432ull + 16777216ull;
    float*  stats = (float*)(wq + 12ull * 1024 * 1024);
    float* mean1 = stats;        float* rinv1 = stats + 512;
    float* mean2 = stats + 1024; float* rinv2 = stats + 1536;

    size_t wcur = 0;
    auto qw = [&](const float* w, int O, int I, int K) -> int8_t* {
        wcur = (wcur + 15) & ~(size_t)15;
        int8_t* dst = (int8_t*)(wq + wcur);
        int total = O * I * K * K;
        wcur += (size_t)total;
        kQuantW<<<dim3((total + 255) / 256), dim3(256), 0, stream>>>(w, dst, O, I, K);
        return dst;
    };
    auto zeroAct = [&](int8_t* a, int H, int W, int C) {
        int c16 = (128 * (H + 2) * (W + 2) * C) / 16;
        kZero16<<<dim3((c16 + 255) / 256), dim3(256), 0, stream>>>(a, c16);
    };

    int8_t* w1s = qw(conv1w, 64, 3, 3);
    {
        int total = 131072 * 64;
        kStem<<<dim3((total + 255) / 256), dim3(256), 0, stream>>>(x, w1s, F1);
        kBNStats<<<dim3(64), dim3(256), 0, stream>>>(F1, 64, 131072, mean1, rinv1);
        zeroAct(actA, 32, 32, 64);
        kBNQuantPad<<<dim3((total + 255) / 256), dim3(256), 0, stream>>>(
            F1, bn1g, bn1b, mean1, rinv1, actA, 64, 32, 32, total);
    }

    int8_t* cur = actA;
    int8_t* nxt = actB;
    int H = 32, W = 32;
    for (int i = 0; i < 8; ++i) {
        const Blk& b = blks[i];
        const int Ho = H / b.st, Wo = W / b.st;
        const int Ncol = 128 * Ho * Wo;
        const int total = Ncol * b.pl;
        const dim3 egrid((total + 255) / 256), eblk(256);
        const dim3 cgrid(Ncol / 64, b.pl / 16), cblk(32);

        int8_t* w1 = qw(b.c1w, b.pl, b.inp, 3);
        int8_t* w2 = qw(b.c2w, b.pl, b.pl, 3);
        int8_t* wS = b.scw ? qw(b.scw, b.pl, b.inp, 1) : nullptr;

        kBinConvWmma<<<cgrid, cblk, 0, stream>>>(
            cur, w1, F1, b.inp, b.pl, H, W, Ho, Wo, b.st, 1, 3);
        kBNStats<<<dim3(b.pl), dim3(256), 0, stream>>>(F1, b.pl, Ncol, mean1, rinv1);
        zeroAct(actH, Ho, Wo, b.pl);
        kBNQuantPad<<<egrid, eblk, 0, stream>>>(
            F1, b.g1, b.b1, mean1, rinv1, actH, b.pl, Ho, Wo, total);

        kBinConvWmma<<<cgrid, cblk, 0, stream>>>(
            actH, w2, F1, b.pl, b.pl, Ho, Wo, Ho, Wo, 1, 1, 3);
        kBNStats<<<dim3(b.pl), dim3(256), 0, stream>>>(F1, b.pl, Ncol, mean1, rinv1);
        kBNApply<<<egrid, eblk, 0, stream>>>(
            F1, b.g2, b.b2, mean1, rinv1, b.pl, total);

        zeroAct(nxt, Ho, Wo, b.pl);
        if (wS) {
            kBinConvWmma<<<cgrid, cblk, 0, stream>>>(
                cur, wS, F2, b.inp, b.pl, H, W, Ho, Wo, b.st, 0, 1);
            kBNStats<<<dim3(b.pl), dim3(256), 0, stream>>>(F2, b.pl, Ncol, mean2, rinv2);
            kBNApply<<<egrid, eblk, 0, stream>>>(
                F2, b.gs, b.bs, mean2, rinv2, b.pl, total);
            kAddQuantFPad<<<egrid, eblk, 0, stream>>>(F1, F2, nxt, b.pl, Ho, Wo, total);
        } else {
            kAddQuantIPad<<<egrid, eblk, 0, stream>>>(F1, cur, nxt, b.pl, Ho, Wo, total);
        }

        int8_t* t = cur; cur = nxt; nxt = t;
        H = Ho; W = Wo;
    }

    kFC<<<dim3((1280 + 63) / 64), dim3(64), 0, stream>>>(cur, fcw, fcb, (float*)d_out);
}
